// IndRNNv2_19069654794464
// MI455X (gfx1250) — matrix operand order repaired and checked by reading, not compiled, not run on hardware
//
#include <hip/hip_runtime.h>

typedef __attribute__((ext_vector_type(4)))  float  v4f;
typedef __attribute__((ext_vector_type(8)))  float  v8f;
typedef __attribute__((ext_vector_type(4)))  __bf16 v4bf;
typedef __attribute__((ext_vector_type(8)))  __bf16 v8bf;
typedef __attribute__((ext_vector_type(16))) __bf16 v16bf;

constexpr int Tn = 1024, Bn = 64, Hn = 512;
constexpr int Mdim = Tn * Bn;   // 65536 rows (t*B+b)
constexpr int Kdim = 512;       // I == H == 512
constexpr int Ndim = 512;

// GEMM tiling: 128x128 workgroup tile, K-chunk 32, 8 waves (4 M x 2 N),
// wave tile 32x64 = 2x4 fragments. Split-bf16 (hi+lo) with f32 accumulate:
// 3x v_wmma_f32_16x16x32_bf16 per 16x16x32 block (hi*hi + hi*lo + lo*hi).
constexpr int BM = 128, BN = 128, KC = 32;
constexpr int LDB = 40;         // LDS row stride in bf16 units (80B): 16B-aligned
                                // b128 frag loads, start banks 20*m mod 64 distinct
constexpr int NTHREADS = 256;

__device__ __forceinline__ v16bf frag16(const __bf16* p0, const __bf16* p1) {
    const v8bf a = *(const v8bf*)p0;
    const v8bf b = *(const v8bf*)p1;
    return __builtin_shufflevector(a, b, 0, 1, 2, 3, 4, 5, 6, 7,
                                   8, 9, 10, 11, 12, 13, 14, 15);
}

// Split W[N][K] f32 into bf16 hi/lo planes once per layer (W is only 1 MB).
__global__ __launch_bounds__(256)
void split_w(const float* __restrict__ W, __bf16* __restrict__ Whi,
             __bf16* __restrict__ Wlo)
{
    const int i = blockIdx.x * 256 + threadIdx.x;   // grid covers N*K exactly
    const float a = W[i];
    const __bf16 h = (__bf16)a;
    Whi[i] = h;
    Wlo[i] = (__bf16)(a - (float)h);
}

__global__ __launch_bounds__(NTHREADS)
void indrnn_gemm(const float*  __restrict__ A,    // [M][K] f32 row-major
                 const __bf16* __restrict__ Whi,  // [N][K] bf16 hi plane
                 const __bf16* __restrict__ Wlo,  // [N][K] bf16 lo plane
                 const float*  __restrict__ bias, // [N]
                 float* __restrict__ C)           // [M][N] f32 row-major
{
    __shared__ __bf16 As_hi[BM * LDB];
    __shared__ __bf16 As_lo[BM * LDB];
    __shared__ __bf16 Ws_hi[BN * LDB];
    __shared__ __bf16 Ws_lo[BN * LDB];

    const int tid   = threadIdx.x;
    const int m0    = blockIdx.y * BM;
    const int n0    = blockIdx.x * BN;
    const int wave  = tid >> 5;
    const int lane  = tid & 31;
    const int lmod  = lane & 15;
    const int lhalf = lane >> 4;
    const int wm    = (wave & 3) * 32;  // wave M offset within tile
    const int wn    = (wave >> 2) * 64; // wave N offset within tile

    v8f acc[2][4];
#pragma unroll
    for (int mi = 0; mi < 2; ++mi)
#pragma unroll
        for (int ni = 0; ni < 4; ++ni)
            acc[mi][ni] = v8f{0.f, 0.f, 0.f, 0.f, 0.f, 0.f, 0.f, 0.f};

    for (int kc = 0; kc < Kdim; kc += KC) {
        // --- A tile: load f32, split to bf16 hi/lo on the fly.
        // 128x32 floats = 1024 quads; 256 threads x 4 iters.
#pragma unroll
        for (int it = 0; it < 4; ++it) {
            const int q  = tid + it * NTHREADS;   // 0..1023
            const int r  = q >> 3;                // tile row 0..127
            const int kq = (q & 7) << 2;          // 0,4,...,28
            const v4f av = *(const v4f*)(A + (size_t)(m0 + r) * Kdim + kc + kq);
            const __bf16 h0 = (__bf16)av.x, h1 = (__bf16)av.y,
                         h2 = (__bf16)av.z, h3 = (__bf16)av.w;
            const v4bf hv = {h0, h1, h2, h3};
            const v4bf lv = {(__bf16)(av.x - (float)h0), (__bf16)(av.y - (float)h1),
                             (__bf16)(av.z - (float)h2), (__bf16)(av.w - (float)h3)};
            *(v4bf*)(&As_hi[r * LDB + kq]) = hv;
            *(v4bf*)(&As_lo[r * LDB + kq]) = lv;
        }
        // --- W tile: pre-split bf16 planes, straight b128 copies.
        // 128x32 bf16 = 512 octets per plane; 256 threads x 2 iters.
#pragma unroll
        for (int it = 0; it < 2; ++it) {
            const int q  = tid + it * NTHREADS;   // 0..511
            const int r  = q >> 2;                // tile row 0..127
            const int ko = (q & 3) << 3;          // 0,8,16,24
            const size_t g = (size_t)(n0 + r) * Kdim + kc + ko;
            *(v8bf*)(&Ws_hi[r * LDB + ko]) = *(const v8bf*)(Whi + g);
            *(v8bf*)(&Ws_lo[r * LDB + ko]) = *(const v8bf*)(Wlo + g);
        }
        __syncthreads();

        // --- Fragments. A 16-bit 16x32 layout: lane half holds K = 8*lhalf+[0..7]
        // and K = 8*lhalf+16+[0..7]. B 32x16 layout: same K = 8*lhalf+[0..7], 8*lhalf+16+[0..7].
        v16bf ah[2], al[2], bh[4], bl[4];
#pragma unroll
        for (int mi = 0; mi < 2; ++mi) {
            const int base = (wm + mi * 16 + lmod) * LDB + 8 * lhalf;
            ah[mi] = frag16(&As_hi[base], &As_hi[base + 16]);
            al[mi] = frag16(&As_lo[base], &As_lo[base + 16]);
        }
#pragma unroll
        for (int ni = 0; ni < 4; ++ni) {
            const int base = (wn + ni * 16 + lmod) * LDB + 8 * lhalf;
            bh[ni] = frag16(&Ws_hi[base], &Ws_hi[base + 16]);
            bl[ni] = frag16(&Ws_lo[base], &Ws_lo[base + 16]);
        }
#pragma unroll
        for (int mi = 0; mi < 2; ++mi)
#pragma unroll
            for (int ni = 0; ni < 4; ++ni) {
                acc[mi][ni] = __builtin_amdgcn_wmma_f32_16x16x32_bf16(
                    false, ah[mi], false, bl[ni], (short)0, acc[mi][ni], false, false);
                acc[mi][ni] = __builtin_amdgcn_wmma_f32_16x16x32_bf16(
                    false, al[mi], false, bh[ni], (short)0, acc[mi][ni], false, false);
                acc[mi][ni] = __builtin_amdgcn_wmma_f32_16x16x32_bf16(
                    false, ah[mi], false, bh[ni], (short)0, acc[mi][ni], false, false);
            }
        __syncthreads();
    }

    // Epilogue: bias add + store. C layout: VGPR j -> M = j + 8*lhalf, N = lmod.
#pragma unroll
    for (int ni = 0; ni < 4; ++ni) {
        const int n  = n0 + wn + ni * 16 + lmod;
        const float bv = bias[n];
#pragma unroll
        for (int mi = 0; mi < 2; ++mi) {
            const int mbase = m0 + wm + mi * 16 + lhalf * 8;
#pragma unroll
            for (int j = 0; j < 8; ++j)
                C[(size_t)(mbase + j) * Ndim + n] = acc[mi][ni][j] + bv;
        }
    }
}

// In-place recurrence: buf[t,c] := h_t = relu(lin[t,c] + u[h]*h_{t-1}), h_{-1}=0.
__global__ __launch_bounds__(256)
void indrnn_scan(float* __restrict__ buf, const float* __restrict__ u,
                 float* __restrict__ hid, int hid_off)
{
    const int BH = Bn * Hn;
    const int c  = blockIdx.x * 256 + threadIdx.x;
    const int b  = c / Hn;
    const int h  = c - b * Hn;
    const float uu = u[h];
    float hv = 0.f;
    float* p = buf + c;
    for (int t = 0; t < Tn; t += 4) {
        const float v0 = p[0];
        const float v1 = p[BH];
        const float v2 = p[2 * BH];
        const float v3 = p[3 * BH];
        hv = fmaxf(fmaf(uu, hv, v0), 0.f); p[0]      = hv;
        hv = fmaxf(fmaf(uu, hv, v1), 0.f); p[BH]     = hv;
        hv = fmaxf(fmaf(uu, hv, v2), 0.f); p[2 * BH] = hv;
        hv = fmaxf(fmaf(uu, hv, v3), 0.f); p[3 * BH] = hv;
        p += 4 * BH;
    }
    hid[b * (2 * Hn) + hid_off + h] = hv;  // concat(hT0, hT1) along last dim
}

extern "C" void kernel_launch(void* const* d_in, const int* in_sizes, int n_in,
                              void* d_out, int out_size, void* d_ws, size_t ws_size,
                              hipStream_t stream)
{
    const float* x  = (const float*)d_in[0];  // (T,B,I)
    const float* W0 = (const float*)d_in[1];  // (H,I)
    const float* b0 = (const float*)d_in[2];
    const float* u0 = (const float*)d_in[3];
    const float* W1 = (const float*)d_in[4];  // (H,H)
    const float* b1 = (const float*)d_in[5];
    const float* u1 = (const float*)d_in[6];

    float* out = (float*)d_out;                       // (T,B,H) output
    float* hid = out + (size_t)Tn * Bn * Hn;          // (B, 2H) hiddens

    // ws layout: [lin: M*N f32 = 128MB][Whi: N*K bf16][Wlo: N*K bf16]
    float*  lin = (float*)d_ws;
    __bf16* whi = (__bf16*)((char*)d_ws + (size_t)Mdim * Ndim * sizeof(float));
    __bf16* wlo = whi + (size_t)Ndim * Kdim;

    dim3 ggrid(Ndim / BN, Mdim / BM);                 // (4, 512)
    const int scan_blocks  = (Bn * Hn) / 256;         // 128
    const int split_blocks = (Ndim * Kdim) / 256;     // 1024

    // Layer 0: lin0 = x @ W0^T + b0 ; scan in place (ws -> ys0), hT0 -> hid[:, 0:H]
    split_w<<<split_blocks, 256, 0, stream>>>(W0, whi, wlo);
    indrnn_gemm<<<ggrid, NTHREADS, 0, stream>>>(x, whi, wlo, b0, lin);
    indrnn_scan<<<scan_blocks, 256, 0, stream>>>(lin, u0, hid, 0);
    // Layer 1: lin1 = ys0 @ W1^T + b1 (into d_out) ; scan in place, hT1 -> hid[:, H:2H]
    split_w<<<split_blocks, 256, 0, stream>>>(W1, whi, wlo);
    indrnn_gemm<<<ggrid, NTHREADS, 0, stream>>>(lin, whi, wlo, b1, out);
    indrnn_scan<<<scan_blocks, 256, 0, stream>>>(out, u1, hid, Hn);
}
